// PatchLoss_15857019257542
// MI455X (gfx1250) — matrix operand order repaired and checked by reading, not compiled, not run on hardware
//
#include <hip/hip_runtime.h>
#include <hip/hip_bf16.h>
#include <hip/hip_fp16.h>

typedef __attribute__((ext_vector_type(16))) _Float16 v16h;
typedef __attribute__((ext_vector_type(8)))  _Float16 v8h;
typedef __attribute__((ext_vector_type(8)))  float    v8f;

// ---- per-level compile-time parameters -------------------------------------
// N      : number of patches (B*4*nh*nh)
// nh     : patches per spatial dim
// st     : stride in gt space
// sh     : pooling scale (1,2,4)
// ce     : crop_emb spatial (pshape//2)
// pst    : pred_stride in feature space
// fH     : feature spatial size
// cbase  : boundary channel base (0 for short mask, 4 for long)
template<int LVL> struct LP;
template<> struct LP<0> { static constexpr int N=2312, nh=17, st=15, sh=1, ce=7,  pst=15, fH=256, cbase=0; };
template<> struct LP<1> { static constexpr int N=1352, nh=13, st=18, sh=2, ce=15, pst=9,  fH=128, cbase=0; };
template<> struct LP<2> { static constexpr int N=968,  nh=11, st=18, sh=4, ce=30, pst=4,  fH=64,  cbase=4; };

#define NPAD 1136           // 1125 padded to 71 tiles of 16
#define NTILES 71

// ---- stage A: gather emb -> f16 [Mpad][32], zero padded rows ---------------
template<int LVL>
__global__ void stage_A_kern(const float* __restrict__ feat, _Float16* __restrict__ Ah) {
  constexpr int N = LP<LVL>::N, nh = LP<LVL>::nh, pst = LP<LVL>::pst, fH = LP<LVL>::fH;
  constexpr int Mpad = ((N + 15) / 16) * 16;
  int idx = blockIdx.x * 256 + threadIdx.x;
  if (idx >= Mpad * 32) return;
  int m = idx >> 5, k = idx & 31;
  float v = 0.f;
  if (m < N) {
    constexpr int nhw = nh * nh;
    int b  = m / (4 * nhw);
    int r  = m % (4 * nhw);
    int d  = r / nhw;
    int r2 = r % nhw;
    int h  = r2 / nh;
    int w  = r2 % nh;
    const size_t fHW = (size_t)fH * fH;
    size_t off = (size_t)b * (32 * 8 * fHW) + (size_t)k * (8 * fHW)
               + (size_t)(2 + d) * fHW + (size_t)(7 + h * pst) * fH + (size_t)(7 + w * pst);
    v = feat[off];
  }
  Ah[idx] = (_Float16)v;
}

// ---- stage W: transpose W(32x1125) -> f16 Wt[1136][32], pad bias -----------
__global__ void stage_W_kern(const float* __restrict__ W, const float* __restrict__ bias,
                             _Float16* __restrict__ Wt, float* __restrict__ biasP) {
  int idx = blockIdx.x * 256 + threadIdx.x;   // NPAD*32 elements
  if (idx >= NPAD * 32) return;
  int n = idx >> 5, k = idx & 31;
  float v = (n < 1125) ? W[(size_t)k * 1125 + n] : 0.f;
  Wt[(size_t)n * 32 + k] = (_Float16)v;
  if (k == 0) biasP[n] = (n < 1125) ? bias[n] : 0.f;
}

// ---- fused WMMA GEMM + sigmoid + target/pool/masks + wave reduction --------
template<int LVL>
__global__ __launch_bounds__(256)
void gemm_loss_kern(const _Float16* __restrict__ Ah, const _Float16* __restrict__ Wt,
                    const float* __restrict__ biasP, const int* __restrict__ gt,
                    const int* __restrict__ bnd, float2* __restrict__ partials) {
  constexpr int N = LP<LVL>::N, nh = LP<LVL>::nh, st = LP<LVL>::st, sh = LP<LVL>::sh,
                ce = LP<LVL>::ce, cbase = LP<LVL>::cbase;
  constexpr int Mtiles = (N + 15) / 16;

  const int wave = threadIdx.x >> 5;
  const int lane = threadIdx.x & 31;
  const int mtile = blockIdx.y * 8 + wave;
  const int ntile = blockIdx.x;            // 0..70
  if (mtile >= Mtiles) return;             // uniform per wave -> EXEC stays all-1s
  const int half = lane >> 4, l = lane & 15;

  // A fragment (lane M=l; K = 16*half .. 16*half+15, contiguous in Ah row)
  const _Float16* arow = Ah + (size_t)(mtile * 16 + l) * 32;
  v16h a = *(const v16h*)(arow + 16 * half);

  // B fragment (lane N=l; K = 16*half .. 16*half+15, contiguous in Wt row)
  const int ncol = ntile * 16 + l;
  v16h bfr = *(const v16h*)(Wt + (size_t)ncol * 32 + 16 * half);

  v8f c = {};
  c = __builtin_amdgcn_wmma_f32_16x16x32_f16(false, a, false, bfr, (short)0, c, false, false);

  // ---- epilogue: per output element (patch m, patch-cell col) ----
  float num = 0.f, den = 0.f;
  const bool colok = ncol < 1125;
  const int cd = ncol / 225, rr = ncol % 225, ci = rr / 15, cj = rr % 15;
  const float biasv = biasP[ncol];

  #pragma unroll
  for (int i = 0; i < 8; ++i) {
    int m = mtile * 16 + i + 8 * half;     // C/D layout: VGPR i holds M = i + 8*half
    if (m < N && colok) {
      constexpr int nhw = nh * nh;
      int b  = m / (4 * nhw);
      int r  = m % (4 * nhw);
      int d  = r / nhw;
      int r2 = r % nhw;
      int h  = r2 / nh;
      int w  = r2 % nh;
      const size_t gbase = (size_t)b * 524288;      // 8*256*256
      // center label + boundary mask at crop_emb position
      int zc = 2 + d, yc = ce + h * st, xc = ce + w * st;
      int cval = gt[gbase + (size_t)zc * 65536 + (size_t)yc * 256 + xc];
      size_t bb = (size_t)b * 4194304 + (size_t)zc * 65536 + (size_t)yc * 256 + xc;
      int onb = 0;
      #pragma unroll
      for (int cc = 0; cc < 4; ++cc)
        onb |= bnd[bb + (size_t)(cbase + cc) * 524288];
      bool valid = (cval != 0) && (onb == 0);
      // pooled target (max of gt!=center) and ignore (any gt==0) over sh x sh
      int z = d + cd;
      int trg = 0, ign = 0;
      #pragma unroll
      for (int pi = 0; pi < sh; ++pi)
        #pragma unroll
        for (int pj = 0; pj < sh; ++pj) {
          int y = h * st + ci * sh + pi;
          int x = w * st + cj * sh + pj;
          int g = gt[gbase + (size_t)z * 65536 + (size_t)y * 256 + x];
          trg |= (g != cval);
          ign |= (g == 0);
        }
      float logit = c[i] + biasv;
      float p = 1.f / (1.f + __expf(-logit));
      float t = (float)trg;
      if (LVL >= 1) { p = 1.f - p; t = 1.f - t; }
      if (ign)    { p = 0.f; t = 0.f; }
      if (!valid) { p = 0.f; t = 0.f; }
      num += p * t;
      den += p * p + t * t;
    }
  }
  // wave32 butterfly reduction (fixed order -> deterministic)
  #pragma unroll
  for (int off = 16; off > 0; off >>= 1) {
    num += __shfl_xor(num, off, 32);
    den += __shfl_xor(den, off, 32);
  }
  if (lane == 0) partials[(size_t)mtile * NTILES + ntile] = make_float2(num, den);
}

// ---- deterministic fixed-tree reduction of per-tile partials ---------------
__global__ void reduce_kern(const float2* __restrict__ partials, int count,
                            float* __restrict__ acc, int lvl) {
  __shared__ float sn[256], sd[256];
  float n = 0.f, d = 0.f;
  for (int i = threadIdx.x; i < count; i += 256) {
    float2 v = partials[i];
    n += v.x; d += v.y;
  }
  sn[threadIdx.x] = n; sd[threadIdx.x] = d;
  __syncthreads();
  for (int s = 128; s > 0; s >>= 1) {
    if (threadIdx.x < s) { sn[threadIdx.x] += sn[threadIdx.x + s]; sd[threadIdx.x] += sd[threadIdx.x + s]; }
    __syncthreads();
  }
  if (threadIdx.x == 0) { acc[2 * lvl] = sn[0]; acc[2 * lvl + 1] = sd[0]; }
}

__global__ void finalize_kern(const float* __restrict__ acc, float* __restrict__ out) {
  if (threadIdx.x == 0) {
    float loss = 0.f;
    for (int l = 0; l < 3; ++l)
      loss += -2.f * acc[2 * l] / fmaxf(acc[2 * l + 1], 1e-6f);
    out[0] = loss;
  }
}

// ---- host-side per-level driver --------------------------------------------
template<int LVL>
static void run_level(const float* feat, const float* W, const float* bias,
                      const int* gt, const int* bnd,
                      _Float16* Ah, _Float16* Wt, float* biasP, float2* partials,
                      float* acc, hipStream_t stream) {
  constexpr int N = LP<LVL>::N;
  constexpr int Mtiles = (N + 15) / 16;
  constexpr int Mpad = Mtiles * 16;
  stage_A_kern<LVL><<<(Mpad * 32 + 255) / 256, 256, 0, stream>>>(feat, Ah);
  stage_W_kern<<<(NPAD * 32 + 255) / 256, 256, 0, stream>>>(W, bias, Wt, biasP);
  gemm_loss_kern<LVL><<<dim3(NTILES, (Mtiles + 7) / 8), 256, 0, stream>>>(Ah, Wt, biasP, gt, bnd, partials);
  reduce_kern<<<1, 256, 0, stream>>>(partials, Mtiles * NTILES, acc, LVL);
}

extern "C" void kernel_launch(void* const* d_in, const int* in_sizes, int n_in,
                              void* d_out, int out_size, void* d_ws, size_t ws_size,
                              hipStream_t stream) {
  const int*   gt    = (const int*)d_in[0];
  const int*   bnd   = (const int*)d_in[1];
  const float* feat0 = (const float*)d_in[2];
  const float* feat1 = (const float*)d_in[3];
  const float* feat2 = (const float*)d_in[4];
  const float* W0 = (const float*)d_in[5];  const float* b0 = (const float*)d_in[6];
  const float* W1 = (const float*)d_in[7];  const float* b1 = (const float*)d_in[8];
  const float* W2 = (const float*)d_in[9];  const float* b2 = (const float*)d_in[10];

  char* ws = (char*)d_ws;
  // layout (bytes): acc[8]f32 @0 | Ah @256 (2320*32 f16) | Wt @148736 (1136*32 f16)
  //                 | biasP @221440 (1136 f32) | partials @226048 (145*71 float2)
  float*    acc      = (float*)ws;
  _Float16* Ah       = (_Float16*)(ws + 256);
  _Float16* Wt       = (_Float16*)(ws + 148736);
  float*    biasP    = (float*)(ws + 221440);
  float2*   partials = (float2*)(ws + 226048);

  run_level<0>(feat0, W0, b0, gt, bnd, Ah, Wt, biasP, partials, acc, stream);
  run_level<1>(feat1, W1, b1, gt, bnd, Ah, Wt, biasP, partials, acc, stream);
  run_level<2>(feat2, W2, b2, gt, bnd, Ah, Wt, biasP, partials, acc, stream);

  finalize_kern<<<1, 32, 0, stream>>>(acc, (float*)d_out);
}
